// DualAttentionLayer_18133351924196
// MI455X (gfx1250) — matrix operand order repaired and checked by reading, not compiled, not run on hardware
//
#include <hip/hip_runtime.h>
#include <cstdint>
#include <cstddef>

// ---------------------------------------------------------------------------
// Problem constants (from reference): B=4, N=4096, E=256, H=8, R=128, X=512
// ---------------------------------------------------------------------------
#define BB     4
#define NN_TOK 4096
#define EE     256
#define HH     8
#define HD     32
#define RR     128
#define XX     512
#define GG     64
#define MM     (BB * NN_TOK)   // 16384 tokens

typedef __attribute__((ext_vector_type(16))) __bf16 v16bf;
typedef __attribute__((ext_vector_type(8)))  float  v8f;

// ---------------------------------------------------------------------------
// bf16 helpers (raw-bit storage in unsigned short buffers)
// ---------------------------------------------------------------------------
static __device__ __forceinline__ float bf2f(unsigned short h) {
    unsigned u = ((unsigned)h) << 16;
    return __builtin_bit_cast(float, u);
}
static __device__ __forceinline__ unsigned short f2bf(float f) {
    unsigned u = __builtin_bit_cast(unsigned, f);
    u += 0x7FFFu + ((u >> 16) & 1u);   // round-to-nearest-even
    return (unsigned short)(u >> 16);
}
// single v_rcp_f32 (~1 ulp) instead of IEEE division expansion
static __device__ __forceinline__ float fast_rcp(float x) {
    return __builtin_amdgcn_rcpf(x);
}

union FragU { uint4 u[2]; v16bf v; };

// A fragment: 16(M) x 32(K) bf16, A row-major, K-contiguous.
// Contiguous-k layout used here (same as B): lanes 0-15 -> M=0..15, kbase=0 ; lanes 16-31 -> kbase=16.
// element e (0..15): K = kbase + e  => one contiguous 16-elem run.
static __device__ __forceinline__ v16bf load_a_frag(const unsigned short* __restrict__ A,
                                                    int lda, int row0, int k0, int lane) {
    int m  = lane & 15;
    int kb = (lane >> 4) << 4;
    const uint4* p = reinterpret_cast<const uint4*>(A + (size_t)(row0 + m) * lda + (k0 + kb));
    FragU f; f.u[0] = p[0]; f.u[1] = p[1];   // 16 contiguous elements
    return f.v;
}

// B fragment: 32(K) x 16(N) bf16 from an N-major ("transposed") weight WT[n][k].
// Contiguous-k layout used here (same as A): lane -> col n = lane&15, kbase = (lane>=16)?16:0, element e: K=kbase+e.
static __device__ __forceinline__ v16bf load_b_frag(const unsigned short* __restrict__ WT,
                                                    int ldw, int col0, int k0, int lane) {
    int n  = lane & 15;
    int kb = (lane >> 4) << 4;
    const uint4* p = reinterpret_cast<const uint4*>(WT + (size_t)(col0 + n) * ldw + (k0 + kb));
    FragU f; f.u[0] = p[0]; f.u[1] = p[1];   // 16 contiguous elements
    return f.v;
}

static __device__ __forceinline__ v8f wmma_bf16(v16bf a, v16bf b, v8f c) {
    return __builtin_amdgcn_wmma_f32_16x16x32_bf16(false, a, false, b, (short)0, c, false, false);
}

// activation codes
#define ACT_NONE  0
#define ACT_RELU6 1
#define ACT_ELU1  2

// ---------------------------------------------------------------------------
// Generic streaming GEMM:  C[M,Nn] = act(A[M,K] @ W[K,Nn] + bias)
// A: bf16 row-major; W passed pre-transposed bf16 [Nn][K]; outputs f32 and/or bf16.
// Block = 128 threads = 4 waves; wave owns a 16-row tile; NT 16-col tiles per wave.
// ---------------------------------------------------------------------------
template<int NT>
__global__ void __launch_bounds__(128)
gemm_bf16_kernel(const unsigned short* __restrict__ A,
                 const unsigned short* __restrict__ WT,
                 const float* __restrict__ bias,
                 float* __restrict__ outF,
                 unsigned short* __restrict__ outB,
                 int K, int Nn, int act) {
    const int wave = threadIdx.x >> 5;
    const int lane = threadIdx.x & 31;
    const int row0 = (blockIdx.x * 4 + wave) * 16;
    const int col0 = blockIdx.y * (NT * 16);

    v8f acc[NT] = {};
    for (int k0 = 0; k0 < K; k0 += 32) {
        v16bf a = load_a_frag(A, K, row0, k0, lane);
#pragma unroll
        for (int nt = 0; nt < NT; ++nt) {
            v16bf b = load_b_frag(WT, K, col0 + nt * 16, k0, lane);
            acc[nt] = wmma_bf16(a, b, acc[nt]);
        }
    }
    const int n  = lane & 15;
    const int mb = (lane >> 4) << 3;
#pragma unroll
    for (int nt = 0; nt < NT; ++nt) {
        int col = col0 + nt * 16 + n;
        float bv = bias ? bias[col] : 0.0f;
#pragma unroll
        for (int v = 0; v < 8; ++v) {
            int row = row0 + mb + v;
            float x = acc[nt][v] + bv;
            if (act == ACT_RELU6)      x = fminf(fmaxf(x, 0.0f), 6.0f);
            else if (act == ACT_ELU1)  x = (x > 0.0f) ? (x + 1.0f) : __expf(x);
            size_t o = (size_t)row * Nn + col;
            if (outF) outF[o] = x;
            if (outB) outB[o] = f2bf(x);
        }
    }
}

// ---------------------------------------------------------------------------
// Weight prep: f32 [K][Nn] -> bf16 N-major [Nn][K]
// ---------------------------------------------------------------------------
__global__ void transposeW_kernel(const float* __restrict__ W, unsigned short* __restrict__ WT,
                                  int K, int Nn) {
    int i = blockIdx.x * 256 + threadIdx.x;
    if (i >= K * Nn) return;
    int n = i / K, k = i % K;
    WT[i] = f2bf(W[(size_t)k * Nn + n]);
}

__global__ void f32_to_bf16_kernel(const float* __restrict__ x, unsigned short* __restrict__ y, int n) {
    int i = blockIdx.x * 256 + threadIdx.x;
    if (i < n) y[i] = f2bf(x[i]);
}

// ---------------------------------------------------------------------------
// ksum[b,e] = sum_n k[b,n,e]   (two-stage, deterministic)
// ---------------------------------------------------------------------------
__global__ void ksum_partial_kernel(const unsigned short* __restrict__ k, float* __restrict__ part) {
    int b = blockIdx.x >> 4, chunk = blockIdx.x & 15, e = threadIdx.x;
    const unsigned short* p = k + ((size_t)(b * NN_TOK) + chunk * 256) * EE + e;
    float acc = 0.0f;
#pragma unroll 4
    for (int n = 0; n < 256; ++n) acc += bf2f(p[(size_t)n * EE]);
    part[(size_t)blockIdx.x * EE + e] = acc;
}
__global__ void ksum_reduce_kernel(const float* __restrict__ part, float* __restrict__ ksum) {
    int i = blockIdx.x * 256 + threadIdx.x;     // i over B*E
    int b = i / EE, e = i % EE;
    float acc = 0.0f;
#pragma unroll
    for (int c = 0; c < 16; ++c) acc += part[(size_t)(b * 16 + c) * EE + e];
    ksum[i] = acc;
}

// ---------------------------------------------------------------------------
// kv[b,h,d,e] = sum_n k[b,n,h,d] * v[b,n,h,e]  -> partials, then reduce to
// kvT bf16 stored [bh][e][d] (N-major for the WMMA B-fragment loader).
// ---------------------------------------------------------------------------
__global__ void kv_partial_kernel(const unsigned short* __restrict__ kbuf,
                                  const unsigned short* __restrict__ vbuf,
                                  float* __restrict__ kvp) {
    __shared__ float kx[64][32];
    __shared__ float vx[64][32];
    int bh = blockIdx.x, b = bh >> 3, h = bh & 7;
    int t = threadIdx.x, d = t >> 3, e4 = (t & 7) * 4;
    float a0 = 0, a1 = 0, a2 = 0, a3 = 0;
    int nbase = b * NN_TOK + blockIdx.y * 512;
    for (int c = 0; c < 512; c += 64) {
#pragma unroll
        for (int i = 0; i < 8; ++i) {
            int idx = i * 256 + t, nn = idx >> 5, dd = idx & 31;
            size_t off = (size_t)(nbase + c + nn) * EE + h * HD + dd;
            kx[nn][dd] = bf2f(kbuf[off]);
            vx[nn][dd] = bf2f(vbuf[off]);
        }
        __syncthreads();
#pragma unroll 8
        for (int nn = 0; nn < 64; ++nn) {
            float kd = kx[nn][d];
            a0 += kd * vx[nn][e4 + 0];
            a1 += kd * vx[nn][e4 + 1];
            a2 += kd * vx[nn][e4 + 2];
            a3 += kd * vx[nn][e4 + 3];
        }
        __syncthreads();
    }
    float* o = kvp + (((size_t)bh * 8 + blockIdx.y) * 32 + d) * 32 + e4;
    o[0] = a0; o[1] = a1; o[2] = a2; o[3] = a3;
}
__global__ void kv_reduce_kernel(const float* __restrict__ kvp, unsigned short* __restrict__ kvT) {
    int i = blockIdx.x * 256 + threadIdx.x;      // over 32*1024
    int bh = i >> 10, r = i & 1023, d = r >> 5, e = r & 31;
    float acc = 0.0f;
#pragma unroll
    for (int p = 0; p < 8; ++p) acc += kvp[(((size_t)bh * 8 + p) << 10) + r];
    kvT[((size_t)bh << 10) + e * 32 + d] = f2bf(acc);   // transposed store
}

// denom[token,h] = sum_d q[token, h*32+d] * ksum[b, h*32+d]
__global__ void denom_kernel(const unsigned short* __restrict__ q,
                             const float* __restrict__ ksum,
                             float* __restrict__ denom) {
    int i = blockIdx.x * 256 + threadIdx.x;      // over M*H
    int token = i >> 3, h = i & 7, b = token >> 12;
    const unsigned short* qp = q + (size_t)token * EE + h * HD;
    const float* kp = ksum + b * EE + h * HD;
    float acc = 0.0f;
#pragma unroll
    for (int d = 0; d < HD; ++d) acc += bf2f(qp[d]) * kp[d];
    denom[i] = acc;
}

// ctx[token, h*32+e] = (q_bh @ kv_bh)[token,e] / (denom + 1e-6)    (WMMA, K=32)
__global__ void __launch_bounds__(128)
attn_apply_kernel(const unsigned short* __restrict__ q,
                  const unsigned short* __restrict__ kvT,
                  const float* __restrict__ denom,
                  unsigned short* __restrict__ ctx) {
    int bh = blockIdx.x, b = bh >> 3, h = bh & 7;
    int wave = threadIdx.x >> 5, lane = threadIdx.x & 31;
    const unsigned short* kvb = kvT + ((size_t)bh << 10);
    v16bf b0 = load_b_frag(kvb, 32, 0, 0, lane);
    v16bf b1 = load_b_frag(kvb, 32, 16, 0, lane);
    int rowbase = b * NN_TOK + blockIdx.y * 256;
    int n = lane & 15, mb = (lane >> 4) << 3;
    for (int mt = wave; mt < 16; mt += 4) {
        int row0 = rowbase + mt * 16;
        v16bf a = load_a_frag(q, EE, row0, h * HD, lane);
        v8f z = {};
        v8f acc0 = wmma_bf16(a, b0, z);
        v8f acc1 = wmma_bf16(a, b1, z);
#pragma unroll
        for (int v = 0; v < 8; ++v) {
            int row = row0 + mb + v;
            float rinv = fast_rcp(denom[(size_t)row * HH + h] + 1e-6f);   // one v_rcp_f32
            ctx[(size_t)row * EE + h * HD + n]      = f2bf(acc0[v] * rinv);
            ctx[(size_t)row * EE + h * HD + 16 + n] = f2bf(acc1[v] * rinv);
        }
    }
}

// ---------------------------------------------------------------------------
// LayerNorm over E=256. mode 0: x=a+b ; mode 1: x = a*g0 + b*g1 (gated combine)
// ---------------------------------------------------------------------------
__global__ void ln_kernel(const float* __restrict__ a, const float* __restrict__ b,
                          const float* __restrict__ gates,
                          const float* __restrict__ gamma, const float* __restrict__ beta,
                          float* __restrict__ outF, unsigned short* __restrict__ outB, int mode) {
    __shared__ float s1[256];
    __shared__ float s2[256];
    int row = blockIdx.x, t = threadIdx.x;
    size_t o = (size_t)row * EE + t;
    float x;
    if (mode == 0) x = a[o] + b[o];
    else {
        float g0 = gates[row * 2 + 0], g1 = gates[row * 2 + 1];
        x = a[o] * g0 + b[o] * g1;
    }
    s1[t] = x; s2[t] = x * x;
    __syncthreads();
    for (int s = 128; s > 0; s >>= 1) {
        if (t < s) { s1[t] += s1[t + s]; s2[t] += s2[t + s]; }
        __syncthreads();
    }
    float mean = s1[0] * (1.0f / EE);
    float var  = s2[0] * (1.0f / EE) - mean * mean;
    float y = (x - mean) * rsqrtf(var + 1e-5f) * gamma[t] + beta[t];
    if (outF) outF[o] = y;
    if (outB) outB[o] = f2bf(y);
}

// concat [M,256]||[M,256] -> [M,512] bf16
__global__ void concat_kernel(const unsigned short* __restrict__ a,
                              const unsigned short* __restrict__ b,
                              unsigned short* __restrict__ out) {
    int i = blockIdx.x * 256 + threadIdx.x;      // over M*512
    int row = i >> 9, c = i & 511;
    out[i] = (c < EE) ? a[(size_t)row * EE + c] : b[(size_t)row * EE + (c - EE)];
}

// 2-way softmax gate: gates[token,{0,1}] = softmax(h[token]@w2 + b2)
__global__ void gate2_kernel(const unsigned short* __restrict__ h,
                             const float* __restrict__ w2, const float* __restrict__ b2,
                             float* __restrict__ gates) {
    int token = blockIdx.x * 256 + threadIdx.x;  // M threads
    const unsigned short* hp = h + (size_t)token * GG;
    float l0 = b2[0], l1 = b2[1];
#pragma unroll 4
    for (int d = 0; d < GG; ++d) {
        float hv = bf2f(hp[d]);
        l0 += hv * w2[d * 2 + 0];
        l1 += hv * w2[d * 2 + 1];
    }
    float m = fmaxf(l0, l1);
    float e0 = __expf(l0 - m), e1 = __expf(l1 - m);
    float inv = fast_rcp(e0 + e1);               // one v_rcp_f32
    gates[(size_t)token * 2 + 0] = e0 * inv;
    gates[(size_t)token * 2 + 1] = e1 * inv;
}

// depthwise conv(k=3, pad 1, per batch) + BatchNorm(eval) + relu6, channels X
__global__ void convbn_kernel(const unsigned short* __restrict__ h1,
                              const float* __restrict__ cw, const float* __restrict__ cb,
                              const float* __restrict__ rm, const float* __restrict__ rv,
                              const float* __restrict__ bng, const float* __restrict__ bnb,
                              unsigned short* __restrict__ hc) {
    int i = blockIdx.x * 256 + threadIdx.x;      // over M*X
    int token = i >> 9, c = i & 511;
    int nloc = token & (NN_TOK - 1);
    float xm = (nloc > 0)            ? bf2f(h1[(size_t)(token - 1) * XX + c]) : 0.0f;
    float xc =                         bf2f(h1[(size_t)token * XX + c]);
    float xp = (nloc < NN_TOK - 1)   ? bf2f(h1[(size_t)(token + 1) * XX + c]) : 0.0f;
    float y = xm * cw[c * 3 + 0] + xc * cw[c * 3 + 1] + xp * cw[c * 3 + 2] + cb[c];
    y = (y - rm[c]) * rsqrtf(rv[c] + 1e-5f) * bng[c] + bnb[c];
    y = fminf(fmaxf(y, 0.0f), 6.0f);
    hc[i] = f2bf(y);
}

// ---------------------------------------------------------------------------
// Host side
// ---------------------------------------------------------------------------
typedef unsigned short bfbits;

struct AttnP { const float *kd,*ku,*kub,*op,*opb,*qd,*qu,*qub,*vd,*vu,*vub; };
struct AttnWT { bfbits *qdT,*quT,*kdT,*kuT,*vdT,*vuT,*opT; };
struct FfnP { const float *b1,*b2,*bnb,*bng,*cb,*cw,*rm,*rv,*w1,*w2; };

static void gemm(hipStream_t s, const bfbits* A, const bfbits* WT, const float* bias,
                 float* outF, bfbits* outB, int K, int Nn, int act) {
    if (Nn % 128 == 0) {
        dim3 grid(MM / 64, Nn / 128);
        gemm_bf16_kernel<8><<<grid, 128, 0, s>>>(A, WT, bias, outF, outB, K, Nn, act);
    } else {
        dim3 grid(MM / 64, Nn / 64);
        gemm_bf16_kernel<4><<<grid, 128, 0, s>>>(A, WT, bias, outF, outB, K, Nn, act);
    }
}
static void prepW(hipStream_t s, const float* W, bfbits* WT, int K, int Nn) {
    int total = K * Nn;
    transposeW_kernel<<<(total + 255) / 256, 256, 0, s>>>(W, WT, K, Nn);
}

struct Scratch {
    bfbits *tB, *qB, *kB, *vB, *ctxB, *kvT;
    float *ksumP, *ksumF, *kvP, *denom;
};

static void run_attn(hipStream_t s, const bfbits* qxB, const bfbits* kxB,
                     const AttnP& p, const AttnWT& w, const Scratch& sc,
                     float* outF, bfbits* outB) {
    gemm(s, qxB, w.qdT, nullptr, nullptr, sc.tB, EE, RR, ACT_NONE);
    gemm(s, sc.tB, w.quT, p.qub, nullptr, sc.qB, RR, EE, ACT_ELU1);
    gemm(s, kxB, w.kdT, nullptr, nullptr, sc.tB, EE, RR, ACT_NONE);
    gemm(s, sc.tB, w.kuT, p.kub, nullptr, sc.kB, RR, EE, ACT_ELU1);
    gemm(s, kxB, w.vdT, nullptr, nullptr, sc.tB, EE, RR, ACT_NONE);
    gemm(s, sc.tB, w.vuT, p.vub, nullptr, sc.vB, RR, EE, ACT_NONE);
    ksum_partial_kernel<<<BB * 16, 256, 0, s>>>(sc.kB, sc.ksumP);
    ksum_reduce_kernel<<<(BB * EE) / 256, 256, 0, s>>>(sc.ksumP, sc.ksumF);
    kv_partial_kernel<<<dim3(BB * HH, 8), 256, 0, s>>>(sc.kB, sc.vB, sc.kvP);
    kv_reduce_kernel<<<(BB * HH * 1024) / 256, 256, 0, s>>>(sc.kvP, sc.kvT);
    denom_kernel<<<(MM * HH) / 256, 256, 0, s>>>(sc.qB, sc.ksumF, sc.denom);
    attn_apply_kernel<<<dim3(BB * HH, 16), 128, 0, s>>>(sc.qB, sc.kvT, sc.denom, sc.ctxB);
    gemm(s, sc.ctxB, w.opT, p.opb, outF, outB, EE, EE, ACT_NONE);
}

// Flattened input indices (JAX pytree sorted-key order):
// 0 body_feats, 1 limb_feats, then params sorted:
// ca_b2l=2, ca_l2b=13, ffn_body=24, ffn_limb=34, gate=44,
// n1b=48, n1l=50, n2b=52, n2l=54, n3b=56, n3l=58, sa_body=60, sa_limb=71
static AttnP attnFrom(void* const* d, int b) {
    AttnP a;
    a.kd  = (const float*)d[b + 0]; a.ku  = (const float*)d[b + 1]; a.kub = (const float*)d[b + 2];
    a.op  = (const float*)d[b + 3]; a.opb = (const float*)d[b + 4];
    a.qd  = (const float*)d[b + 5]; a.qu  = (const float*)d[b + 6]; a.qub = (const float*)d[b + 7];
    a.vd  = (const float*)d[b + 8]; a.vu  = (const float*)d[b + 9]; a.vub = (const float*)d[b + 10];
    return a;
}
static FfnP ffnFrom(void* const* d, int b) {
    FfnP f;
    f.b1 = (const float*)d[b + 0]; f.b2 = (const float*)d[b + 1];
    f.bnb = (const float*)d[b + 2]; f.bng = (const float*)d[b + 3];
    f.cb = (const float*)d[b + 4]; f.cw = (const float*)d[b + 5];
    f.rm = (const float*)d[b + 6]; f.rv = (const float*)d[b + 7];
    f.w1 = (const float*)d[b + 8]; f.w2 = (const float*)d[b + 9];
    return f;
}

extern "C" void kernel_launch(void* const* d_in, const int* in_sizes, int n_in,
                              void* d_out, int out_size, void* d_ws, size_t ws_size,
                              hipStream_t stream) {
    if (!d_in || !d_out || !d_ws || n_in < 82) return;
    hipStream_t s = stream;

    const float* bodyF = (const float*)d_in[0];
    const float* limbF = (const float*)d_in[1];
    AttnP pSaB = attnFrom(d_in, 60), pSaL = attnFrom(d_in, 71);
    AttnP pCbl = attnFrom(d_in, 2),  pClb = attnFrom(d_in, 13);
    FfnP  fB = ffnFrom(d_in, 24),    fL = ffnFrom(d_in, 34);
    const float* gW1 = (const float*)d_in[44 + 2];
    const float* gB1 = (const float*)d_in[44 + 0];
    const float* gW2 = (const float*)d_in[44 + 3];
    const float* gB2 = (const float*)d_in[44 + 1];
    const float* nBeta[6]  = { (const float*)d_in[48], (const float*)d_in[50], (const float*)d_in[52],
                               (const float*)d_in[54], (const float*)d_in[56], (const float*)d_in[58] };
    const float* nGamma[6] = { (const float*)d_in[49], (const float*)d_in[51], (const float*)d_in[53],
                               (const float*)d_in[55], (const float*)d_in[57], (const float*)d_in[59] };

    // ---- workspace bump allocator -------------------------------------
    char* base = (char*)d_ws;
    size_t off = 0;
    auto alloc = [&](size_t bytes) -> void* {
        void* p = base + off;
        off += (bytes + 255) & ~(size_t)255;
        return p;
    };

    // weights (persistent, ~3.1 MB, L2-resident)
    AttnWT wAll[4];
    AttnP  pAll[4] = { pSaB, pSaL, pCbl, pClb };
    for (int i = 0; i < 4; ++i) {
        wAll[i].qdT = (bfbits*)alloc(EE * RR * 2); wAll[i].quT = (bfbits*)alloc(RR * EE * 2);
        wAll[i].kdT = (bfbits*)alloc(EE * RR * 2); wAll[i].kuT = (bfbits*)alloc(RR * EE * 2);
        wAll[i].vdT = (bfbits*)alloc(EE * RR * 2); wAll[i].vuT = (bfbits*)alloc(RR * EE * 2);
        wAll[i].opT = (bfbits*)alloc(EE * EE * 2);
    }
    bfbits* w1T[2]; bfbits* w2T[2];
    w1T[0] = (bfbits*)alloc(EE * XX * 2); w2T[0] = (bfbits*)alloc(XX * EE * 2);
    w1T[1] = (bfbits*)alloc(EE * XX * 2); w2T[1] = (bfbits*)alloc(XX * EE * 2);
    bfbits* gW1T = (bfbits*)alloc(2 * EE * GG * 2);

    // bf16 copies of the input features (freed logically after self-attn)
    bfbits* bodyB = (bfbits*)alloc((size_t)MM * EE * 2);
    bfbits* limbB = (bfbits*)alloc((size_t)MM * EE * 2);
    char* reg_bl = (char*)bodyB;                         // 16 MB, reused for concat

    // attention scratch (~38 MB, reused by all 4 calls, then for body2)
    char* reg_scratch = base + off;
    Scratch sc;
    sc.tB    = (bfbits*)alloc((size_t)MM * RR * 2);
    sc.qB    = (bfbits*)alloc((size_t)MM * EE * 2);
    sc.kB    = (bfbits*)alloc((size_t)MM * EE * 2);
    sc.vB    = (bfbits*)alloc((size_t)MM * EE * 2);
    sc.ksumP = (float*)alloc((size_t)BB * 16 * EE * 4);
    sc.ksumF = (float*)alloc((size_t)BB * EE * 4);
    sc.kvP   = (float*)alloc((size_t)BB * HH * 8 * 1024 * 4);
    sc.kvT   = (bfbits*)alloc((size_t)BB * HH * 1024 * 2);
    sc.denom = (float*)alloc((size_t)MM * HH * 4);
    sc.ctxB  = (bfbits*)alloc((size_t)MM * EE * 2);

    float* saF_b = (float*)alloc((size_t)MM * EE * 4);   // 32 MB region, reused for limb2
    float* saF_l = (float*)alloc((size_t)MM * EE * 4);
    char* reg_sa = (char*)saF_b;

    float*  body1F = (float*)alloc((size_t)MM * EE * 4); // 48 MB region, reused for FFN bufs
    bfbits* body1B = (bfbits*)alloc((size_t)MM * EE * 2);
    float*  limb1F = (float*)alloc((size_t)MM * EE * 4);
    bfbits* limb1B = (bfbits*)alloc((size_t)MM * EE * 2);
    char* reg_b1 = (char*)body1F;

    float*  crossF_b = (float*)alloc((size_t)MM * EE * 4);
    bfbits* crossB_b = (bfbits*)alloc((size_t)MM * EE * 2);
    float*  crossF_l = (float*)alloc((size_t)MM * EE * 4);
    bfbits* crossB_l = (bfbits*)alloc((size_t)MM * EE * 2);

    float*  gatesB = (float*)alloc((size_t)MM * 2 * 4);
    float*  gatesL = (float*)alloc((size_t)MM * 2 * 4);
    bfbits* gatehB = (bfbits*)alloc((size_t)MM * GG * 2);

    // reuses (regions dead by the time these are written):
    bfbits* concatB = (bfbits*)reg_bl;                                   // 16 MB
    float*  body2F  = (float*)reg_scratch;                               // 16 MB
    bfbits* body2B  = (bfbits*)(reg_scratch + (size_t)MM * EE * 4);      //  8 MB
    float*  limb2F  = (float*)reg_sa;                                    // 16 MB
    bfbits* limb2B  = (bfbits*)(reg_sa + (size_t)MM * EE * 4);           //  8 MB
    bfbits* h1B     = (bfbits*)reg_b1;                                   // 16 MB
    bfbits* hcB     = (bfbits*)(reg_b1 + (size_t)MM * XX * 2);           // 16 MB
    float*  ffnoutF = (float*)(reg_b1 + 2 * (size_t)MM * XX * 2);        // 16 MB

    float* outBody = (float*)d_out;
    float* outLimb = (float*)d_out + (size_t)MM * EE;

    // ---- 0) weight prep + input conversion ----------------------------
    for (int i = 0; i < 4; ++i) {
        prepW(s, pAll[i].qd, wAll[i].qdT, EE, RR); prepW(s, pAll[i].qu, wAll[i].quT, RR, EE);
        prepW(s, pAll[i].kd, wAll[i].kdT, EE, RR); prepW(s, pAll[i].ku, wAll[i].kuT, RR, EE);
        prepW(s, pAll[i].vd, wAll[i].vdT, EE, RR); prepW(s, pAll[i].vu, wAll[i].vuT, RR, EE);
        prepW(s, pAll[i].op, wAll[i].opT, EE, EE);
    }
    prepW(s, fB.w1, w1T[0], EE, XX); prepW(s, fB.w2, w2T[0], XX, EE);
    prepW(s, fL.w1, w1T[1], EE, XX); prepW(s, fL.w2, w2T[1], XX, EE);
    prepW(s, gW1, gW1T, 2 * EE, GG);
    f32_to_bf16_kernel<<<(MM * EE) / 256, 256, 0, s>>>(bodyF, bodyB, MM * EE);
    f32_to_bf16_kernel<<<(MM * EE) / 256, 256, 0, s>>>(limbF, limbB, MM * EE);

    // ---- 1) self-attention + LN1 --------------------------------------
    run_attn(s, bodyB, bodyB, pSaB, wAll[0], sc, saF_b, nullptr);
    run_attn(s, limbB, limbB, pSaL, wAll[1], sc, saF_l, nullptr);
    ln_kernel<<<MM, 256, 0, s>>>(bodyF, saF_b, nullptr, nGamma[0], nBeta[0], body1F, body1B, 0);
    ln_kernel<<<MM, 256, 0, s>>>(limbF, saF_l, nullptr, nGamma[1], nBeta[1], limb1F, limb1B, 0);

    // ---- 2) cross-attention -------------------------------------------
    run_attn(s, body1B, limb1B, pCbl, wAll[2], sc, crossF_b, crossB_b);
    run_attn(s, limb1B, body1B, pClb, wAll[3], sc, crossF_l, crossB_l);

    // ---- 3) gates ------------------------------------------------------
    concat_kernel<<<(MM * 2 * EE) / 256, 256, 0, s>>>(body1B, crossB_b, concatB);
    gemm(s, concatB, gW1T, gB1, nullptr, gatehB, 2 * EE, GG, ACT_RELU6);
    gate2_kernel<<<MM / 256, 256, 0, s>>>(gatehB, gW2, gB2, gatesB);
    concat_kernel<<<(MM * 2 * EE) / 256, 256, 0, s>>>(limb1B, crossB_l, concatB);
    gemm(s, concatB, gW1T, gB1, nullptr, gatehB, 2 * EE, GG, ACT_RELU6);
    gate2_kernel<<<MM / 256, 256, 0, s>>>(gatehB, gW2, gB2, gatesL);

    // ---- 4) gated combine + LN2 ---------------------------------------
    ln_kernel<<<MM, 256, 0, s>>>(body1F, crossF_b, gatesB, nGamma[2], nBeta[2], body2F, body2B, 1);
    ln_kernel<<<MM, 256, 0, s>>>(limb1F, crossF_l, gatesL, nGamma[3], nBeta[3], limb2F, limb2B, 1);

    // ---- 5) FFN + LN3 (body, then limb; FFN buffers reused) -----------
    gemm(s, body2B, w1T[0], fB.b1, nullptr, h1B, EE, XX, ACT_RELU6);
    convbn_kernel<<<(MM * XX) / 256, 256, 0, s>>>(h1B, fB.cw, fB.cb, fB.rm, fB.rv, fB.bng, fB.bnb, hcB);
    gemm(s, hcB, w2T[0], fB.b2, ffnoutF, nullptr, XX, EE, ACT_NONE);
    ln_kernel<<<MM, 256, 0, s>>>(body2F, ffnoutF, nullptr, nGamma[4], nBeta[4], outBody, nullptr, 0);

    gemm(s, limb2B, w1T[1], fL.b1, nullptr, h1B, EE, XX, ACT_RELU6);
    convbn_kernel<<<(MM * XX) / 256, 256, 0, s>>>(h1B, fL.cw, fL.cb, fL.rm, fL.rv, fL.bng, fL.bnb, hcB);
    gemm(s, hcB, w2T[1], fL.b2, ffnoutF, nullptr, XX, EE, ACT_NONE);
    ln_kernel<<<MM, 256, 0, s>>>(limb2F, ffnoutF, nullptr, nGamma[5], nBeta[5], outLimb, nullptr, 0);

    (void)in_sizes; (void)out_size; (void)ws_size;
}
